// BitNetAttention_76828374990912
// MI455X (gfx1250) — 16-bit and 8-bit matrix operand order repaired and checked by reading, compiled, not run on hardware
//
#include <hip/hip_runtime.h>
#include <cstdint>


typedef __attribute__((ext_vector_type(8)))  int       v8i;
typedef __attribute__((ext_vector_type(8)))  float     v8f;
typedef __attribute__((ext_vector_type(16))) _Float16  v16h;

#define B_  2
#define S_  2048
#define H_  2048
#define NH_ 16
#define HD_ 128
#define LD_ 64
#define M_  (B_ * S_)
#define ML_ (B_ * S_ * NH_)


__device__ inline v8i load_a_i8(const int8_t* __restrict__ base, int ldk, int lane) {
    const int m  = lane & 15;
    const int kh = (lane >> 4) & 1;
    const int8_t* row = base + (size_t)m * ldk + kh * 8;
    union { uint2 u2[4]; v8i v; } u;
#pragma unroll
    for (int c = 0; c < 4; ++c)
        u.u2[c] = *reinterpret_cast<const uint2*>(row + 16 * c);
    return u.v;
}

__device__ inline v8i load_b_i8(const int8_t* __restrict__ base, int ldk, int lane) {
    const int n  = lane & 15;
    const int kh = (lane >> 4) & 1;
    const int8_t* row = base + (size_t)n * ldk + kh * 8;
    union { uint2 u2[4]; v8i v; } u;
#pragma unroll
    for (int c = 0; c < 4; ++c)
        u.u2[c] = *reinterpret_cast<const uint2*>(row + 16 * c);
    return u.v;
}

__device__ inline v16h load_a_f16(const _Float16* __restrict__ base, int ldk, int lane) {
    const int m  = lane & 15;
    const int kh = (lane >> 4) & 1;
    const _Float16* row = base + (size_t)m * ldk + kh * 8;
    union { uint4 u4[2]; v16h v; } u;
    u.u4[0] = *reinterpret_cast<const uint4*>(row);
    u.u4[1] = *reinterpret_cast<const uint4*>(row + 16);
    return u.v;
}

__device__ inline v16h load_b_f16(const _Float16* __restrict__ base, int ldk, int lane) {
    const int n  = lane & 15;
    const int kh = (lane >> 4) & 1;
    const _Float16* row = base + (size_t)n * ldk + kh * 8;
    union { uint4 u4[2]; v16h v; } u;
    u.u4[0] = *reinterpret_cast<const uint4*>(row);
    u.u4[1] = *reinterpret_cast<const uint4*>(row + 16);
    return u.v;
}


__global__ __launch_bounds__(1024)
void k_weight_scale(const float* __restrict__ w, int n, float* __restrict__ out) {
    __shared__ float red[1024];
    float s = 0.f;
    for (int i = threadIdx.x; i < n; i += 1024) s += fabsf(w[i]);
    red[threadIdx.x] = s;
    __syncthreads();
    for (int off = 512; off > 0; off >>= 1) {
        if ((int)threadIdx.x < off) red[threadIdx.x] += red[threadIdx.x + off];
        __syncthreads();
    }
    if (threadIdx.x == 0) out[0] = 1.0f / fmaxf(red[0] / (float)n, 1e-5f);
}

__global__ __launch_bounds__(256)
void k_quant_weight(const float* __restrict__ w, const float* __restrict__ scp,
                    int8_t* __restrict__ w8, int n) {
    int i = blockIdx.x * 256 + threadIdx.x;
    if (i >= n) return;
    float q = rintf(w[i] * scp[0]);
    q = fminf(fmaxf(q, -1.f), 1.f);
    w8[i] = (int8_t)q;
}


__device__ inline float block_red(float v, float* red, bool is_max) {
    red[threadIdx.x] = v;
    __syncthreads();
    for (int off = 128; off > 0; off >>= 1) {
        if ((int)threadIdx.x < off)
            red[threadIdx.x] = is_max ? fmaxf(red[threadIdx.x], red[threadIdx.x + off])
                                      : red[threadIdx.x] + red[threadIdx.x + off];
        __syncthreads();
    }
    float r = red[0];
    __syncthreads();
    return r;
}

__global__ __launch_bounds__(256)
void k_subln_quant(const float* __restrict__ x, const float* __restrict__ g,
                   int8_t* __restrict__ xq, float* __restrict__ scl, int rl) {
    __shared__ float red[256];
    const float* xr = x + (size_t)blockIdx.x * rl;
    float s = 0.f, ss = 0.f;
    for (int i = threadIdx.x; i < rl; i += 256) { float v = xr[i]; s += v; ss += v * v; }
    s  = block_red(s, red, false);
    ss = block_red(ss, red, false);
    const float mean = s / rl;
    const float var  = ss / rl - mean * mean;
    const float rstd = rsqrtf(var + 1e-6f);
    float amax = 0.f;
    for (int i = threadIdx.x; i < rl; i += 256) {
        float xn = (xr[i] - mean) * rstd * g[i];
        amax = fmaxf(amax, fabsf(xn));
    }
    amax = block_red(amax, red, true);
    const float sc = 127.f / fmaxf(amax, 1e-5f);
    if (threadIdx.x == 0) scl[blockIdx.x] = sc;
    int8_t* qr = xq + (size_t)blockIdx.x * rl;
    for (int i = threadIdx.x; i < rl; i += 256) {
        float xn = (xr[i] - mean) * rstd * g[i];
        int q = (int)rintf(xn * sc);
        q = q < -128 ? -128 : (q > 127 ? 127 : q);
        qr[i] = (int8_t)q;
    }
}


__global__ __launch_bounds__(128)
void k_gemm_i8(const int8_t* __restrict__ x8, const float* __restrict__ xs,
               const int8_t* __restrict__ w8, const float* __restrict__ wsp,
               float* __restrict__ out, int M, int N, int K) {
    const int lane = threadIdx.x & 31;
    const int wv   = threadIdx.x >> 5;
    const int i0   = (blockIdx.x * 4 + wv) * 16;
    const int j0   = blockIdx.y * 64;
    const int8_t* arow = x8 + (size_t)i0 * K;
    const int8_t* b0p  = w8 + (size_t)(j0 +  0) * K;
    const int8_t* b1p  = w8 + (size_t)(j0 + 16) * K;
    const int8_t* b2p  = w8 + (size_t)(j0 + 32) * K;
    const int8_t* b3p  = w8 + (size_t)(j0 + 48) * K;

    v8i a  = load_a_i8(arow, K, lane);
    v8i b0 = load_b_i8(b0p,  K, lane);
    v8i b1 = load_b_i8(b1p,  K, lane);
    v8i b2 = load_b_i8(b2p,  K, lane);
    v8i b3 = load_b_i8(b3p,  K, lane);

    v8i acc[4] = {};
    for (int k = 0; k < K; k += 64) {
        const int kn = (k + 64 < K) ? k + 64 : k;
        __builtin_prefetch(arow + kn + 64, 0, 3);
        v8i an  = load_a_i8(arow + kn, K, lane);
        v8i bn0 = load_b_i8(b0p + kn,  K, lane);
        v8i bn1 = load_b_i8(b1p + kn,  K, lane);
        v8i bn2 = load_b_i8(b2p + kn,  K, lane);
        v8i bn3 = load_b_i8(b3p + kn,  K, lane);
        acc[0] = __builtin_amdgcn_wmma_i32_16x16x64_iu8(true, a, true, b0, acc[0], false, false);
        acc[1] = __builtin_amdgcn_wmma_i32_16x16x64_iu8(true, a, true, b1, acc[1], false, false);
        acc[2] = __builtin_amdgcn_wmma_i32_16x16x64_iu8(true, a, true, b2, acc[2], false, false);
        acc[3] = __builtin_amdgcn_wmma_i32_16x16x64_iu8(true, a, true, b3, acc[3], false, false);
        a = an; b0 = bn0; b1 = bn1; b2 = bn2; b3 = bn3;
    }

    const float wsc  = wsp[0];
    const int   half = (lane >> 4) & 1;
    const int   c    = lane & 15;
#pragma unroll
    for (int r = 0; r < 8; ++r) {
        const int   row = i0 + r + half * 8;
        const float inv = 1.f / (xs[row] * wsc);
#pragma unroll
        for (int t = 0; t < 4; ++t)
            out[(size_t)row * N + j0 + 16 * t + c] = (float)acc[t][r] * inv;
    }
}

__global__ __launch_bounds__(128)
void k_gemm_lat(const int8_t* __restrict__ x8, const float* __restrict__ xs,
                const int8_t* __restrict__ w8, const float* __restrict__ wsp,
                _Float16* __restrict__ out) {
    const int K = HD_;
    const int lane = threadIdx.x & 31;
    const int wv   = threadIdx.x >> 5;
    const int i0   = (blockIdx.x * 4 + wv) * 16;
    const int8_t* arow = x8 + (size_t)i0 * K;
    v8i acc[4] = {};
#pragma unroll
    for (int k = 0; k < K; k += 64) {
        v8i a = load_a_i8(arow + k, K, lane);
#pragma unroll
        for (int t = 0; t < 4; ++t) {
            v8i b = load_b_i8(w8 + (size_t)(16 * t) * K + k, K, lane);
            acc[t] = __builtin_amdgcn_wmma_i32_16x16x64_iu8(true, a, true, b, acc[t], false, false);
        }
    }
    const float wsc  = wsp[0];
    const int   half = (lane >> 4) & 1;
    const int   c    = lane & 15;
#pragma unroll
    for (int r = 0; r < 8; ++r) {
        const int row = i0 + r + half * 8;
        const int h  = row & (NH_ - 1);
        const int bs = row >> 4;
        const int s  = bs & (S_ - 1);
        const int b  = bs >> 11;
        const float inv = 1.f / (xs[row] * wsc);
        _Float16* orow = out + ((((size_t)b * NH_ + h) * S_) + s) * LD_;
#pragma unroll
        for (int t = 0; t < 4; ++t)
            orow[16 * t + c] = (_Float16)((float)acc[t][r] * inv);
    }
}

__global__ __launch_bounds__(256)
void k_rope(float* __restrict__ q, float* __restrict__ k, int npairs) {
    int idx = blockIdx.x * 256 + threadIdx.x;
    if (idx >= npairs) return;
    const int d = idx & 63;
    const size_t base = (size_t)(idx >> 6) * HD_;
    const float f = __expf(-0.14391156f * (float)d);
    float q1 = q[base + d], q2 = q[base + d + 64];
    q[base + d]      = -q2 * f;
    q[base + d + 64] =  q1 * f;
    float k1 = k[base + d], k2 = k[base + d + 64];
    k[base + d]      = -k2 * f;
    k[base + d + 64] =  k1 * f;
}

__global__ __launch_bounds__(256)
void k_v_to_f16T(const float* __restrict__ v, _Float16* __restrict__ vT, int total) {
    int i = blockIdx.x * 256 + threadIdx.x;
    if (i >= total) return;
    int hd = i & (HD_ - 1);
    int r  = i >> 7;
    int h  = r & (NH_ - 1);
    int r2 = r >> 4;
    int s  = r2 & (S_ - 1);
    int b  = r2 >> 11;
    vT[(((size_t)b * NH_ + h) * HD_ + hd) * S_ + s] = (_Float16)v[i];
}

__global__ __launch_bounds__(256)
void k_flash(const _Float16* __restrict__ qlat, const _Float16* __restrict__ klat,
             const _Float16* __restrict__ vT, float* __restrict__ out) {
    __shared__ __align__(16) _Float16 sP[8][16][32];
    const int lane  = threadIdx.x & 31;
    const int wslot = threadIdx.x >> 5;
    const int wave  = blockIdx.x * 8 + wslot;
    const int mt    = wave & (S_ / 16 - 1);
    const int bh    = wave >> 7;
    const int m0    = mt * 16;

    const _Float16* qb = qlat + ((size_t)bh * S_ + m0) * LD_;
    const _Float16* kb = klat + (size_t)bh * S_ * LD_;
    const _Float16* vb = vT   + (size_t)bh * HD_ * S_;

    const v16h aq0 = load_a_f16(qb,      LD_, lane);
    const v16h aq1 = load_a_f16(qb + 32, LD_, lane);

    v8f acc[8] = {};
    float mrow[8], lrow[8];
#pragma unroll
    for (int r = 0; r < 8; ++r) { mrow[r] = -1e30f; lrow[r] = 0.f; }

    v16h bk0 = load_b_f16(kb,            LD_, lane);
    v16h bk1 = load_b_f16(kb + 32,       LD_, lane);
    v16h bk2 = load_b_f16(kb + 16 * LD_,      LD_, lane);
    v16h bk3 = load_b_f16(kb + 16 * LD_ + 32, LD_, lane);

    const int half = (lane >> 4) & 1;
    for (int t0 = 0; t0 < S_; t0 += 32) {
        v8f c0 = {}, c1 = {};
        c0 = __builtin_amdgcn_wmma_f32_16x16x32_f16(false, aq0, false, bk0, (short)0, c0, false, false);
        c0 = __builtin_amdgcn_wmma_f32_16x16x32_f16(false, aq1, false, bk1, (short)0, c0, false, false);
        c1 = __builtin_amdgcn_wmma_f32_16x16x32_f16(false, aq0, false, bk2, (short)0, c1, false, false);
        c1 = __builtin_amdgcn_wmma_f32_16x16x32_f16(false, aq1, false, bk3, (short)0, c1, false, false);

        const int tn = (t0 + 32 < S_) ? t0 + 32 : t0;
        const _Float16* kn = kb + (size_t)tn * LD_;
        v16h nk0 = load_b_f16(kn,            LD_, lane);
        v16h nk1 = load_b_f16(kn + 32,       LD_, lane);
        v16h nk2 = load_b_f16(kn + 16 * LD_,      LD_, lane);
        v16h nk3 = load_b_f16(kn + 16 * LD_ + 32, LD_, lane);

#pragma unroll
        for (int r = 0; r < 8; ++r) {
            float s0 = c0[r] * 0.125f;
            float s1 = c1[r] * 0.125f;
            float mx = fmaxf(s0, s1);
            mx = fmaxf(mx, __shfl_xor(mx, 1));
            mx = fmaxf(mx, __shfl_xor(mx, 2));
            mx = fmaxf(mx, __shfl_xor(mx, 4));
            mx = fmaxf(mx, __shfl_xor(mx, 8));
            float mnew  = fmaxf(mrow[r], mx);
            float alpha = __expf(mrow[r] - mnew);
            float p0 = __expf(s0 - mnew);
            float p1 = __expf(s1 - mnew);
            float rs = p0 + p1;
            rs += __shfl_xor(rs, 1);
            rs += __shfl_xor(rs, 2);
            rs += __shfl_xor(rs, 4);
            rs += __shfl_xor(rs, 8);
            lrow[r] = lrow[r] * alpha + rs;
            mrow[r] = mnew;
#pragma unroll
            for (int j = 0; j < 8; ++j) acc[j][r] *= alpha;
            const int row = r + half * 8;
            const int col = lane & 15;
            sP[wslot][row][col]      = (_Float16)p0;
            sP[wslot][row][col + 16] = (_Float16)p1;
        }
        __syncthreads();
        const v16h ap = load_a_f16(&sP[wslot][0][0], 32, lane);
        v16h bv = load_b_f16(vb + t0, S_, lane);
#pragma unroll
        for (int j = 0; j < 8; ++j) {
            v16h bvn = bv;
            if (j < 7) bvn = load_b_f16(vb + (size_t)((j + 1) * 16) * S_ + t0, S_, lane);
            acc[j] = __builtin_amdgcn_wmma_f32_16x16x32_f16(false, ap, false, bv, (short)0, acc[j], false, false);
            bv = bvn;
        }
        __syncthreads();
        bk0 = nk0; bk1 = nk1; bk2 = nk2; bk3 = nk3;
    }

    const int b = bh >> 4, h = bh & (NH_ - 1);
#pragma unroll
    for (int r = 0; r < 8; ++r) {
        const int   row = m0 + r + half * 8;
        const float inv = 1.f / lrow[r];
#pragma unroll
        for (int j = 0; j < 8; ++j) {
            int col = h * HD_ + j * 16 + (lane & 15);
            out[((size_t)b * S_ + row) * H_ + col] = acc[j][r] * inv;
        }
    }
}


extern "C" void kernel_launch(void* const* d_in, const int* in_sizes, int n_in,
                              void* d_out, int out_size, void* d_ws, size_t ws_size,
                              hipStream_t stream) {
    const float* hs  = (const float*)d_in[0];
    const float* wq  = (const float*)d_in[1];
    const float* gq  = (const float*)d_in[2];
    const float* wk  = (const float*)d_in[3];
    const float* gk  = (const float*)d_in[4];
    const float* wv  = (const float*)d_in[5];
    const float* gv  = (const float*)d_in[6];
    const float* wo  = (const float*)d_in[7];
    const float* go  = (const float*)d_in[8];
    const float* wlq = (const float*)d_in[9];
    const float* glq = (const float*)d_in[10];
    const float* wlk = (const float*)d_in[11];
    const float* glk = (const float*)d_in[12];
    float* out = (float*)d_out;

    char*  ws  = (char*)d_ws;
    size_t off = 0;
    auto carve = [&](size_t bytes) -> void* {
        void* p = ws + off;
        off = (off + bytes + 255) & ~(size_t)255;
        return p;
    };
    float*     wsc   = (float*)    carve(8 * sizeof(float));
    int8_t*    x8    = (int8_t*)   carve((size_t)M_ * H_);
    float*     xs    = (float*)    carve((size_t)M_ * 4);
    int8_t*    wq8   = (int8_t*)   carve((size_t)H_ * H_);
    int8_t*    wk8   = (int8_t*)   carve((size_t)H_ * H_);
    int8_t*    wv8   = (int8_t*)   carve((size_t)H_ * H_);
    int8_t*    wo8   = (int8_t*)   carve((size_t)H_ * H_);
    int8_t*    wlq8  = (int8_t*)   carve((size_t)LD_ * HD_);
    int8_t*    wlk8  = (int8_t*)   carve((size_t)LD_ * HD_);
    float*     qf    = (float*)    carve((size_t)M_ * H_ * 4);
    float*     kf    = (float*)    carve((size_t)M_ * H_ * 4);
    float*     vf    = (float*)    carve((size_t)M_ * H_ * 4);
    _Float16*  vT16  = (_Float16*) carve((size_t)M_ * H_ * 2);
    int8_t*    lq8   = (int8_t*)   carve((size_t)ML_ * HD_);
    float*     lqs   = (float*)    carve((size_t)ML_ * 4);
    int8_t*    lk8   = (int8_t*)   carve((size_t)ML_ * HD_);
    float*     lks   = (float*)    carve((size_t)ML_ * 4);
    _Float16*  qlat  = (_Float16*) carve((size_t)ML_ * LD_ * 2);
    _Float16*  klat  = (_Float16*) carve((size_t)ML_ * LD_ * 2);
    float*     aout  = qf;
    int8_t*    o8    = x8;
    float*     os    = xs;

    k_weight_scale<<<1, 1024, 0, stream>>>(wq,  H_ * H_,   wsc + 0);
    k_weight_scale<<<1, 1024, 0, stream>>>(wk,  H_ * H_,   wsc + 1);
    k_weight_scale<<<1, 1024, 0, stream>>>(wv,  H_ * H_,   wsc + 2);
    k_weight_scale<<<1, 1024, 0, stream>>>(wo,  H_ * H_,   wsc + 3);
    k_weight_scale<<<1, 1024, 0, stream>>>(wlq, LD_ * HD_, wsc + 4);
    k_weight_scale<<<1, 1024, 0, stream>>>(wlk, LD_ * HD_, wsc + 5);
    const int nw  = H_ * H_;
    const int nwl = LD_ * HD_;
    k_quant_weight<<<(nw  + 255) / 256, 256, 0, stream>>>(wq,  wsc + 0, wq8,  nw);
    k_quant_weight<<<(nw  + 255) / 256, 256, 0, stream>>>(wk,  wsc + 1, wk8,  nw);
    k_quant_weight<<<(nw  + 255) / 256, 256, 0, stream>>>(wv,  wsc + 2, wv8,  nw);
    k_quant_weight<<<(nw  + 255) / 256, 256, 0, stream>>>(wo,  wsc + 3, wo8,  nw);
    k_quant_weight<<<(nwl + 255) / 256, 256, 0, stream>>>(wlq, wsc + 4, wlq8, nwl);
    k_quant_weight<<<(nwl + 255) / 256, 256, 0, stream>>>(wlk, wsc + 5, wlk8, nwl);

    dim3 gBig(M_ / 64, H_ / 64);
    k_subln_quant<<<M_, 256, 0, stream>>>(hs, gq, x8, xs, H_);
    k_gemm_i8<<<gBig, 128, 0, stream>>>(x8, xs, wq8, wsc + 0, qf, M_, H_, H_);
    k_subln_quant<<<M_, 256, 0, stream>>>(hs, gk, x8, xs, H_);
    k_gemm_i8<<<gBig, 128, 0, stream>>>(x8, xs, wk8, wsc + 1, kf, M_, H_, H_);
    k_subln_quant<<<M_, 256, 0, stream>>>(hs, gv, x8, xs, H_);
    k_gemm_i8<<<gBig, 128, 0, stream>>>(x8, xs, wv8, wsc + 2, vf, M_, H_, H_);

    const int npairs = ML_ * 64;
    k_rope<<<(npairs + 255) / 256, 256, 0, stream>>>(qf, kf, npairs);
    const int nvel = ML_ * HD_;
    k_v_to_f16T<<<(nvel + 255) / 256, 256, 0, stream>>>(vf, vT16, nvel);

    k_subln_quant<<<ML_, 256, 0, stream>>>(qf, glq, lq8, lqs, HD_);
    k_subln_quant<<<ML_, 256, 0, stream>>>(kf, glk, lk8, lks, HD_);
    k_gemm_lat<<<ML_ / 64, 128, 0, stream>>>(lq8, lqs, wlq8, wsc + 4, qlat);
    k_gemm_lat<<<ML_ / 64, 128, 0, stream>>>(lk8, lks, wlk8, wsc + 5, klat);

    k_flash<<<(B_ * NH_ * (S_ / 16)) / 8, 256, 0, stream>>>(qlat, klat, vT16, aout);

    k_subln_quant<<<M_, 256, 0, stream>>>(aout, go, o8, os, H_);
    k_gemm_i8<<<gBig, 128, 0, stream>>>(o8, os, wo8, wsc + 3, out, M_, H_, H_);

    (void)in_sizes; (void)n_in; (void)out_size; (void)ws_size;
}
